// StatefulCausalAttentionHead_34342558498819
// MI455X (gfx1250) — matrix operand order repaired and checked by reading, not compiled, not run on hardware
//
#include <hip/hip_runtime.h>
#include <hip/hip_fp16.h>
#include <cmath>

typedef __attribute__((ext_vector_type(16))) _Float16 v16h;
typedef __attribute__((ext_vector_type(8)))  float    v8f;
typedef __attribute__((ext_vector_type(4)))  unsigned v4u;
typedef __attribute__((ext_vector_type(8)))  unsigned v8u;

// Problem constants (B=2, S=256, L=2048 -> T=2560, D=512, DK=DV=64)
#define TT      2560
#define NB      2
#define DMODEL  512
#define HD      64
#define NTILE   160         // TT / 16
#define SEG1    256
#define SEG2    2304

// ---------------------------------------------------------------------------
// WMMA helpers — A and B both use contiguous k order k = 16*(lane/16) + i (wave32)
// ---------------------------------------------------------------------------
__device__ __forceinline__ v8f wmma16(v16h a, v16h b, v8f c) {
  return __builtin_amdgcn_wmma_f32_16x16x32_f16(
      /*neg_a=*/false, a, /*neg_b=*/false, b,
      /*c_mod=*/(short)0, c, /*reuse_a=*/false, /*reuse_b=*/false);
}

// A-matrix 16x32 f16 fragment from a row-major f16 buffer (rowStride halves).
__device__ __forceinline__ v16h load_a_frag(const _Float16* base, int rowStride,
                                            int kbase, int lane) {
  int m  = lane & 15;
  int hi = (lane & 16) ? 16 : 0;
  const _Float16* rp = base + m * rowStride + kbase + hi;
  v16h a;
#pragma unroll
  for (int g = 0; g < 8; ++g) {
    int k0 = 2 * g;
    a[2 * g]     = rp[k0];
    a[2 * g + 1] = rp[k0 + 1];
  }
  return a;
}

// B-matrix 32x16 f16 fragment where each logical N-column is a contiguous
// K-run in memory (e.g. K16 rows of length 64).
__device__ __forceinline__ v16h load_b_frag_rows(const _Float16* base,
                                                 int rowStride, int kbase,
                                                 int lane) {
  int n  = lane & 15;
  int d0 = kbase + ((lane >> 4) << 4);
  const _Float16* p = base + n * rowStride + d0;  // 16 contiguous halves
  v16h b;
#pragma unroll
  for (int i = 0; i < 16; ++i) b[i] = p[i];
  return b;
}

// B-matrix fragment from a K-major LDS tile tile[K][16].
__device__ __forceinline__ v16h load_b_frag_tile(const _Float16* tile,
                                                 int kbase, int lane) {
  int n  = lane & 15;
  int d0 = kbase + ((lane >> 4) << 4);
  v16h b;
#pragma unroll
  for (int i = 0; i < 16; ++i) b[i] = tile[(d0 + i) * 16 + n];
  return b;
}

// B-matrix fragment from an f32 weight matrix W[K][64] (K-major, row len 64).
__device__ __forceinline__ v16h load_b_frag_w(const float* W, int n0, int kbase,
                                              int lane) {
  int n  = n0 + (lane & 15);
  int d0 = kbase + ((lane >> 4) << 4);
  v16h b;
#pragma unroll
  for (int i = 0; i < 16; ++i) b[i] = (_Float16)W[(d0 + i) * HD + n];
  return b;
}

__device__ __forceinline__ float sigmoidf_(float x) {
  return 1.0f / (1.0f + __expf(-x));
}

// ---------------------------------------------------------------------------
// gfx1250 async / TDM data movement (CDNA5 ISA)
// ---------------------------------------------------------------------------
// Generic (shared-aperture) pointer -> 32-bit LDS offset: per ISA flat->LDS
// mapping, the LDS offset is the low 32 address bits.
__device__ __forceinline__ unsigned lds_off_of(const void* p) {
  return (unsigned)(uintptr_t)p;
}
// WAR guard: ensure outstanding ds_loads consumed their LDS data before the
// async engine overwrites the staging tile.
__device__ __forceinline__ void lds_war_fence() {
  asm volatile("s_wait_dscnt 0x0" ::: "memory");
}
// 16B global -> LDS, no VGPR transit, tracked on ASYNCcnt.
__device__ __forceinline__ void async_b128(unsigned lds_off, const void* gsrc) {
  asm volatile("global_load_async_to_lds_b128 %0, %1, off"
               :: "v"(lds_off), "v"(gsrc)
               : "memory");
}
__device__ __forceinline__ void async_wait0() {
  asm volatile("s_wait_asynccnt 0x0" ::: "memory");
}
// TDM: 1D tensor load (nelem 2-byte elements, contiguous) into LDS.
// D# group0: count=1, lds_addr, global_addr, type=2 ("image").
// D# group1: data_size=2B, tensor_dim0=tile_dim0=nelem, tensor_dim1=tile_dim1=1.
__device__ __forceinline__ void tdm_load_1d_f16(unsigned lds_off,
                                                const void* gsrc,
                                                unsigned nelem) {
  unsigned long long ga = (unsigned long long)(uintptr_t)gsrc;
  v4u g0;
  g0[0] = 1u;                                               // count=1
  g0[1] = lds_off;                                          // lds_addr
  g0[2] = (unsigned)ga;                                     // global_addr lo
  g0[3] = (unsigned)((ga >> 32) & 0x01FFFFFFull) | 0x80000000u;  // hi | type=2
  v8u g1;
  g1[0] = 0x00010000u;                                      // data_size=1 (2B)
  g1[1] = (nelem & 0xFFFFu) << 16;                          // tensor_dim0 lo16
  g1[2] = ((nelem >> 16) & 0xFFFFu) | (1u << 16);           // dim0 hi | dim1=1
  g1[3] = (nelem & 0xFFFFu) << 16;                          // tile_dim0
  g1[4] = 1u;                                               // tile_dim1=1
  g1[5] = nelem;                                            // dim0_stride lo
  g1[6] = 0u;
  g1[7] = 0u;
  asm volatile("tensor_load_to_lds %0, %1" :: "s"(g0), "s"(g1) : "memory");
}

// ---------------------------------------------------------------------------
// Kernel 0: per-row LayerNorm (segment-specific params) + Q/K/V projection.
// ---------------------------------------------------------------------------
__global__ __launch_bounds__(256) void ln_qkv_kernel(
    const float* __restrict__ x,
    const float* __restrict__ wq_ss, const float* __restrict__ wq_m,
    const float* __restrict__ wq_se, const float* __restrict__ wk_ss,
    const float* __restrict__ wk_m,  const float* __restrict__ wk_se,
    const float* __restrict__ wv_ss, const float* __restrict__ wv_m,
    const float* __restrict__ wv_se, const float* __restrict__ g_ss,
    const float* __restrict__ b_ss,  const float* __restrict__ g_m,
    const float* __restrict__ b_m,   const float* __restrict__ g_se,
    const float* __restrict__ b_se,
    _Float16* __restrict__ Q16, _Float16* __restrict__ K16,
    _Float16* __restrict__ V16) {
  __shared__ _Float16 xn[16 * DMODEL];  // 16 KB of normalized rows (f16)

  const int tid  = threadIdx.x;
  const int lane = tid & 31;
  const int w    = tid >> 5;
  const int bid  = blockIdx.x;
  const int b    = bid / NTILE;
  const int t0   = (bid % NTILE) * 16;

  const int seg = (t0 < SEG1) ? 0 : ((t0 < SEG2) ? 1 : 2);
  const float* g  = (seg == 0) ? g_ss : ((seg == 1) ? g_m : g_se);
  const float* bb = (seg == 0) ? b_ss : ((seg == 1) ? b_m : b_se);
  const float* Wq = (seg == 0) ? wq_ss : ((seg == 1) ? wq_m : wq_se);
  const float* Wk = (seg == 0) ? wk_ss : ((seg == 1) ? wk_m : wk_se);
  const float* Wv = (seg == 0) ? wv_ss : ((seg == 1) ? wv_m : wv_se);

  // --- LayerNorm: each wave handles 2 rows (16 elems / lane / row) ---
#pragma unroll
  for (int rr = 0; rr < 2; ++rr) {
    const int r = w * 2 + rr;
    const float* xr = x + ((size_t)b * TT + t0 + r) * DMODEL;
    float vals[16];
    float s = 0.0f;
#pragma unroll
    for (int e = 0; e < 16; ++e) { vals[e] = xr[lane + 32 * e]; s += vals[e]; }
#pragma unroll
    for (int off = 16; off > 0; off >>= 1) s += __shfl_xor(s, off, 32);
    const float mean = s * (1.0f / (float)DMODEL);
    float vv = 0.0f;
#pragma unroll
    for (int e = 0; e < 16; ++e) { float d = vals[e] - mean; vv += d * d; }
#pragma unroll
    for (int off = 16; off > 0; off >>= 1) vv += __shfl_xor(vv, off, 32);
    const float rstd = rsqrtf(vv * (1.0f / (float)DMODEL) + 1e-5f);
#pragma unroll
    for (int e = 0; e < 16; ++e) {
      const int c = lane + 32 * e;
      xn[r * DMODEL + c] = (_Float16)((vals[e] - mean) * rstd * g[c] + bb[c]);
    }
  }
  __syncthreads();

  // --- Projections: 12 (output, n-tile) tasks over 8 waves ---
  for (int task = w; task < 12; task += 8) {
    const int o  = task >> 2;           // 0=Q 1=K 2=V
    const int n0 = (task & 3) * 16;
    const float* W = (o == 0) ? Wq : ((o == 1) ? Wk : Wv);
    _Float16* dst  = (o == 0) ? Q16 : ((o == 1) ? K16 : V16);
    v8f acc = {};
#pragma unroll 4
    for (int kk = 0; kk < DMODEL / 32; ++kk) {
      v16h a  = load_a_frag(xn, DMODEL, kk * 32, lane);
      v16h bf = load_b_frag_w(W, n0, kk * 32, lane);
      acc = wmma16(a, bf, acc);
    }
    const int N  = lane & 15;
    const int Mb = (lane >> 4) * 8;
#pragma unroll
    for (int r = 0; r < 8; ++r)
      dst[((size_t)b * TT + t0 + Mb + r) * HD + n0 + N] = (_Float16)acc[r];
  }
}

// ---------------------------------------------------------------------------
// Kernel 0b: cope_emb f32 -> f16
// ---------------------------------------------------------------------------
__global__ __launch_bounds__(256) void cope_to_h_kernel(
    const float* __restrict__ src, _Float16* __restrict__ dst, int n) {
  int i = blockIdx.x * 256 + threadIdx.x;
  if (i < n) dst[i] = (_Float16)src[i];
}

// ---------------------------------------------------------------------------
// Kernel 1: fused attention. One WG = 16 query rows, 8 waves, big dynamic LDS.
// ---------------------------------------------------------------------------
// Dynamic-LDS layout (bytes):
#define OFF_LROW 0u                            // float   [16*TT] = 163840
#define OFF_LIP  (OFF_LROW + 16u * TT * 4u)    // _Float16[16*TT] =  81920
#define OFF_QA   (OFF_LIP + 16u * TT * 2u)     // _Float16[16*64] =   2048
#define OFF_CT   (OFF_QA + 16u * 64u * 2u)     // _Float16[8][64*16] = 16384
#define OFF_PART (OFF_CT + 8u * 64u * 16u * 2u)// float   [256]   =   1024
#define OFF_ORED (OFF_PART + 256u * 4u)        // float   [4*32*8]=   4096
#define SMEM_TOTAL (OFF_ORED + 4u * 32u * 8u * 4u)   // 269312 bytes

__global__ __launch_bounds__(256) void attn_kernel(
    const _Float16* __restrict__ Q16, const _Float16* __restrict__ K16,
    const _Float16* __restrict__ V16, const _Float16* __restrict__ C16,
    float* __restrict__ out) {
  extern __shared__ char smem[];
  float*    Lrow = (float*)(smem + OFF_LROW);     // att logits rows (f32)
  _Float16* liP  = (_Float16*)(smem + OFF_LIP);   // li rows, later P rows
  _Float16* qa   = (_Float16*)(smem + OFF_QA);    // 16x64 q tile
  _Float16* ctA  = (_Float16*)(smem + OFF_CT);    // per-wave staging tiles
  float*    part = (float*)(smem + OFF_PART);     // scan / reduce scratch
  float*    ored = (float*)(smem + OFF_ORED);     // cross-wave C reduction

  const int tid  = threadIdx.x;
  const int lane = tid & 31;
  const int w    = tid >> 5;
  const int bid  = blockIdx.x;
  const int b    = bid / NTILE;
  const int t0   = (bid % NTILE) * 16;

  // --- Stage 16x64 Q tile via the Tensor Data Mover (wave 0 issues DMA) ---
  if (w == 0) {
    tdm_load_1d_f16(lds_off_of(qa), Q16 + ((size_t)b * TT + t0) * HD, 16 * HD);
    __builtin_amdgcn_s_wait_tensorcnt(0);
  }
  __syncthreads();

  const v16h a0 = load_a_frag(qa, HD, 0, lane);
  const v16h a1 = load_a_frag(qa, HD, 32, lane);
  const int  N  = lane & 15;
  const int  Mb = (lane >> 4) * 8;

  // --- Phase 1: logits rows  L[i][s] = q[t0+i] . k[s]  (f32 accum) ---
#pragma unroll 2
  for (int st = w; st < NTILE; st += 8) {
    const int s0 = st * 16;
    const _Float16* kb = K16 + ((size_t)b * TT + s0) * HD;
    if (st + 8 < NTILE)
      __builtin_prefetch(K16 + ((size_t)b * TT + s0 + 128) * HD, 0, 1);
    v16h b0 = load_b_frag_rows(kb, HD, 0, lane);
    v16h b1 = load_b_frag_rows(kb, HD, 32, lane);
    v8f acc = {};
    acc = wmma16(a0, b0, acc);
    acc = wmma16(a1, b1, acc);
#pragma unroll
    for (int r = 0; r < 8; ++r) Lrow[(Mb + r) * TT + s0 + N] = acc[r];
  }

  // --- Phase 2: li rows  li[i][m] = q[t0+i] . cope[:,m] ---
  _Float16* ct = ctA + w * 1024;  // per-wave 64x16 staging tile
  const unsigned ct_off = lds_off_of(ct) + (unsigned)lane * 64u;
  for (int mt = w; mt < NTILE; mt += 8) {
    const int m0 = mt * 16;
    // async-stage cope[d][m0..m0+15], d=0..63: each lane 2 d-rows x 32B
    const _Float16* cb = C16 + (size_t)(2 * lane) * TT + m0;
    lds_war_fence();                 // prior gathers done before overwrite
    async_b128(ct_off,      cb);
    async_b128(ct_off + 16, cb + 8);
    async_b128(ct_off + 32, cb + TT);
    async_b128(ct_off + 48, cb + TT + 8);
    async_wait0();
    v16h b0 = load_b_frag_tile(ct, 0, lane);
    v16h b1 = load_b_frag_tile(ct, 32, lane);
    v8f acc = {};
    acc = wmma16(a0, b0, acc);
    acc = wmma16(a1, b1, acc);
#pragma unroll
    for (int r = 0; r < 8; ++r)
      liP[(Mb + r) * TT + m0 + N] = (_Float16)acc[r];
  }
  __syncthreads();

  // --- Phase 3: per-row suffix scan of sigmoid(L) + CoPE bias + causal ---
  const int CH = TT / 256;  // 10 elements per thread per row
  for (int i = 0; i < 16; ++i) {
    float* Li = Lrow + i * TT;
    const _Float16* lii = liP + i * TT;
    const int t = t0 + i;
    const int sbase = tid * CH;
    float c = 0.0f;
#pragma unroll
    for (int e = 0; e < CH; ++e) c += sigmoidf_(Li[sbase + e]);
    part[tid] = c;
    __syncthreads();
    // Hillis-Steele inclusive suffix scan over 256 chunk sums
    for (int off = 1; off < 256; off <<= 1) {
      float add = (tid + off < 256) ? part[tid + off] : 0.0f;
      __syncthreads();
      part[tid] += add;
      __syncthreads();
    }
    float running = part[tid] - c;  // exclusive suffix (chunks above mine)
    for (int e = CH - 1; e >= 0; --e) {
      const int s = sbase + e;
      const float lg = Li[s];
      running += sigmoidf_(lg);
      float pos = fminf(running, (float)(TT - 1));
      float pf  = floorf(pos);
      float wt  = pos - pf;
      int ipf = (int)pf;
      int ipc = (int)ceilf(pos);
      float bias = (float)lii[ipc] * wt + (float)lii[ipf] * (1.0f - wt);
      Li[s] = lg * 0.125f + ((s <= t) ? bias : -INFINITY);
    }
    __syncthreads();
  }

  // --- Phase 4: softmax per row, store P as f16 into liP ---
  for (int i = 0; i < 16; ++i) {
    float* Li = Lrow + i * TT;
    float mx = -INFINITY;
    for (int s = tid; s < TT; s += 256) mx = fmaxf(mx, Li[s]);
    part[tid] = mx;
    __syncthreads();
    for (int off = 128; off > 0; off >>= 1) {
      if (tid < off) part[tid] = fmaxf(part[tid], part[tid + off]);
      __syncthreads();
    }
    const float rmax = part[0];
    __syncthreads();
    float sm = 0.0f;
    for (int s = tid; s < TT; s += 256) sm += __expf(Li[s] - rmax);
    part[tid] = sm;
    __syncthreads();
    for (int off = 128; off > 0; off >>= 1) {
      if (tid < off) part[tid] += part[tid + off];
      __syncthreads();
    }
    const float inv = 1.0f / part[0];
    __syncthreads();
    for (int s = tid; s < TT; s += 256)
      liP[i * TT + s] = (_Float16)(__expf(Li[s] - rmax) * inv);
  }
  __syncthreads();

  // --- Phase 5: out = P @ V  (waves split 4 n-tiles x 2 K-halves) ---
  const int ntile = w & 3;
  const int khalf = w >> 2;
  const int n0 = ntile * 16;
  _Float16* vt = ctA + w * 1024;  // reuse per-wave tile (32x16 used)
  const unsigned vt_off = lds_off_of(vt) + (unsigned)lane * 32u;
  v8f acc = {};
  const int sb = khalf * (TT / 2);
  for (int blk = 0; blk < TT / 64; ++blk) {  // 40 blocks of K=32
    const int s0 = sb + blk * 32;
    // async-stage V[s0+lane][n0..n0+15] (32B/lane) into the wave tile
    const _Float16* vsrc = V16 + ((size_t)b * TT + s0 + lane) * HD + n0;
    lds_war_fence();
    async_b128(vt_off,      vsrc);
    async_b128(vt_off + 16, vsrc + 8);
    async_wait0();
    v16h a  = load_a_frag(liP, TT, s0, lane);
    v16h bf = load_b_frag_tile(vt, 0, lane);
    acc = wmma16(a, bf, acc);
  }
  if (khalf == 1) {
    float* dp = ored + (ntile * 32 + lane) * 8;
#pragma unroll
    for (int r = 0; r < 8; ++r) dp[r] = acc[r];
  }
  __syncthreads();
  if (khalf == 0) {
    const float* sp = ored + (ntile * 32 + lane) * 8;
#pragma unroll
    for (int r = 0; r < 8; ++r)
      out[((size_t)b * TT + t0 + Mb + r) * HD + n0 + N] = acc[r] + sp[r];
  }
}

// ---------------------------------------------------------------------------
extern "C" void kernel_launch(void* const* d_in, const int* in_sizes, int n_in,
                              void* d_out, int out_size, void* d_ws,
                              size_t ws_size, hipStream_t stream) {
  (void)in_sizes; (void)n_in; (void)out_size; (void)ws_size;
  const float* x     = (const float*)d_in[0];
  const float* w_q   = (const float*)d_in[1];
  const float* w_k   = (const float*)d_in[2];
  const float* w_v   = (const float*)d_in[3];
  const float* w_qss = (const float*)d_in[4];
  const float* w_kss = (const float*)d_in[5];
  const float* w_vss = (const float*)d_in[6];
  const float* w_qse = (const float*)d_in[7];
  const float* w_kse = (const float*)d_in[8];
  const float* w_vse = (const float*)d_in[9];
  const float* ln_g    = (const float*)d_in[10];
  const float* ln_b    = (const float*)d_in[11];
  const float* ln_g_ss = (const float*)d_in[12];
  const float* ln_b_ss = (const float*)d_in[13];
  const float* ln_g_se = (const float*)d_in[14];
  const float* ln_b_se = (const float*)d_in[15];
  const float* cope    = (const float*)d_in[16];

  // Workspace: f16 Q/K/V (B*T*64 each) + f16 cope (64*T). ~2.2 MB total.
  const size_t qkv_elems = (size_t)NB * TT * HD;
  _Float16* Q16 = (_Float16*)d_ws;
  _Float16* K16 = Q16 + qkv_elems;
  _Float16* V16 = K16 + qkv_elems;
  _Float16* C16 = V16 + qkv_elems;

  const int cope_n = HD * TT;
  cope_to_h_kernel<<<(cope_n + 255) / 256, 256, 0, stream>>>(cope, C16, cope_n);

  ln_qkv_kernel<<<NB * NTILE, 256, 0, stream>>>(
      x, w_qss, w_q, w_qse, w_kss, w_k, w_kse, w_vss, w_v, w_vse,
      ln_g_ss, ln_b_ss, ln_g, ln_b, ln_g_se, ln_b_se, Q16, K16, V16);

  hipFuncSetAttribute(reinterpret_cast<const void*>(attn_kernel),
                      hipFuncAttributeMaxDynamicSharedMemorySize,
                      (int)SMEM_TOTAL);
  attn_kernel<<<NB * NTILE, 256, SMEM_TOTAL, stream>>>(Q16, K16, V16, C16,
                                                       (float*)d_out);
}
